// CNN_OTAM_CLIPFSAR_43095701848304
// MI455X (gfx1250) — matrix operand order repaired and checked by reading, not compiled, not run on hardware
//
#include <hip/hip_runtime.h>
#include <hip/hip_bf16.h>
#include <math.h>

// ---------------------------------------------------------------------------
// OTAM (soft-DTW) bidirectional logits, fused on gfx1250:
//   Phase 1: L2-normalize frame features, split each fp32 into bf16 hi+lo.
//   Phase 2: per (q,n) pair: 16x16 cosine-sim tile via split-bf16 WMMA
//            (hi*hi + hi*lo + lo*hi, f32 accumulate ~= fp32 GEMM accuracy),
//            then run both DTW directions inside the wave (anti-diagonal
//            wavefront, lanes 0-15 forward / 16-31 transposed, shfl_up deps).
// ---------------------------------------------------------------------------

typedef __bf16 bf16_t;
typedef __attribute__((ext_vector_type(16))) __bf16 v16bf;
typedef __attribute__((ext_vector_type(8)))  __bf16 v8bf;
typedef __attribute__((ext_vector_type(8)))  float  v8f;

#define DDIM   1024
#define QV     512          // query videos
#define NV     128          // support videos
#define LFRM   16           // frames per video
#define LBDA   0.5f

__device__ __forceinline__ v8f wmma_bf16(v16bf a, v16bf b, v8f c) {
  return __builtin_amdgcn_wmma_f32_16x16x32_bf16(
      /*neg_a=*/false, a, /*neg_b=*/false, b,
      /*c_mod=*/(short)0, c, /*reuse_a=*/false, /*reuse_b=*/false);
}

// A-operand (16x32 bf16, M x K): lane m = lane&15, half = lane>>4.
// elems 0..15 <- K = k0 .. k0+15 contiguous (k0 already includes +16*half)
__device__ __forceinline__ v16bf load_a_frag(const bf16_t* __restrict__ row, int k0) {
  return *(const v16bf*)(row + k0);
}

// B-operand (32x16 bf16, K x N): lane n = lane&15, half = lane>>4.
// elems 0..15 <- K = k0 .. k0+15 contiguous (k0 already includes +16*half)
__device__ __forceinline__ v16bf load_b_frag(const bf16_t* __restrict__ row, int k0) {
  return *(const v16bf*)(row + k0);
}

// ---------------------------------------------------------------------------
// Kernel 1: per-frame L2 normalize + split into bf16 hi/lo pair.
// One 256-thread block per frame row of 1024 floats.
// Rows [0,2048)  -> support, rows [2048,10240) -> target.
// ---------------------------------------------------------------------------
__global__ __launch_bounds__(256)
void normalize_split_kernel(const float* __restrict__ support,
                            const float* __restrict__ target,
                            bf16_t* __restrict__ tnh, bf16_t* __restrict__ tnl,
                            bf16_t* __restrict__ snh, bf16_t* __restrict__ snl) {
  const int tid = threadIdx.x;
  const int row = blockIdx.x;

  const float* src;
  bf16_t *dh, *dl;
  if (row < NV * LFRM) {
    src = support + (size_t)row * DDIM;
    dh  = snh + (size_t)row * DDIM;
    dl  = snl + (size_t)row * DDIM;
  } else {
    const int r = row - NV * LFRM;
    src = target + (size_t)r * DDIM;
    dh  = tnh + (size_t)r * DDIM;
    dl  = tnl + (size_t)r * DDIM;
  }

  float x[4];
  float ss = 0.0f;
#pragma unroll
  for (int i = 0; i < 4; ++i) {
    x[i] = src[tid + 256 * i];
    ss += x[i] * x[i];
  }

  __shared__ float red[256];
  red[tid] = ss;
  __syncthreads();
#pragma unroll
  for (int s = 128; s > 0; s >>= 1) {
    if (tid < s) red[tid] += red[tid + s];
    __syncthreads();
  }
  const float inv = rsqrtf(red[0]);

#pragma unroll
  for (int i = 0; i < 4; ++i) {
    const float v = x[i] * inv;
    const bf16_t h = (bf16_t)v;            // RTNE
    const float  rem = v - (float)h;       // residual captured in lo term
    dh[tid + 256 * i] = h;
    dl[tid + 256 * i] = (bf16_t)rem;
  }
}

// ---------------------------------------------------------------------------
// Kernel 2: fused WMMA similarity + bidirectional OTAM DP.
// Block = 128 threads = 4 waves. Wave w handles pairs (q, n0) and (q, n0+1),
// q = blockIdx.y, n0 = blockIdx.x*8 + 2*w.
// ---------------------------------------------------------------------------
__global__ __launch_bounds__(128)
void otam_fused_kernel(const bf16_t* __restrict__ tnh, const bf16_t* __restrict__ tnl,
                       const bf16_t* __restrict__ snh, const bf16_t* __restrict__ snl,
                       float* __restrict__ out) {
  const int lane  = threadIdx.x & 31;
  const int wid   = threadIdx.x >> 5;
  const int lrow  = lane & 15;
  const int khalf = lane >> 4;

  const int q  = blockIdx.y;
  const int n0 = blockIdx.x * 8 + wid * 2;

  const size_t rowA  = ((size_t)q  * LFRM + lrow) * DDIM;
  const size_t rowB0 = ((size_t)n0 * LFRM + lrow) * DDIM;
  const size_t rowB1 = rowB0 + (size_t)LFRM * DDIM;

  const int aoff = khalf * 16;   // A layout: half-wave K offset
  const int boff = khalf * 16;   // B layout: half-wave K offset

  v8f c0 = {};
  v8f c1 = {};

#pragma unroll 2
  for (int kb = 0; kb < DDIM; kb += 32) {
    const v16bf ah  = load_a_frag(tnh + rowA,  kb + aoff);
    const v16bf al  = load_a_frag(tnl + rowA,  kb + aoff);
    const v16bf bh0 = load_b_frag(snh + rowB0, kb + boff);
    const v16bf bl0 = load_b_frag(snl + rowB0, kb + boff);
    const v16bf bh1 = load_b_frag(snh + rowB1, kb + boff);
    const v16bf bl1 = load_b_frag(snl + rowB1, kb + boff);

    // split-bf16 product: hi*hi + hi*lo + lo*hi  (lo*lo ~ 2^-16, dropped)
    c0 = wmma_bf16(ah, bh0, c0);
    c0 = wmma_bf16(ah, bl0, c0);
    c0 = wmma_bf16(al, bh0, c0);

    c1 = wmma_bf16(ah, bh1, c1);
    c1 = wmma_bf16(ah, bl1, c1);
    c1 = wmma_bf16(al, bh1, c1);
  }

  // Stash dists = 1 - sim into this wave's private LDS patch.
  // C layout: VGPR r -> row (r + 8*khalf), col = lane&15.
  __shared__ float ldsD[4][2][LFRM * LFRM];
#pragma unroll
  for (int r = 0; r < 8; ++r) {
    const int lr = r + 8 * khalf;
    ldsD[wid][0][lr * LFRM + lrow] = 1.0f - c0[r];
    ldsD[wid][1][lr * LFRM + lrow] = 1.0f - c1[r];
  }
  __syncthreads();

  // OTAM DP on padded 16 x 18 grid, both directions at once:
  //   lanes 0-15: forward (rows of dists), lanes 16-31: transposed.
  // Anti-diagonal wavefront: at step t, logical row l computes column m=t-l.
  //   needs cum[l-1][m-1] (lane l-1, step t-2)  -> uprev2
  //         cum[l-1][m]   (lane l-1, step t-1)  -> uprev
  //         cum[l][m-1]   (self,     step t-1)  -> left
  const bool fwd = (khalf == 0);
  const int  l   = lrow;

  for (int j = 0; j < 2; ++j) {
    const float* __restrict__ D = ldsD[wid][j];
    float cur = 0.0f, left = 0.0f, uprev = 0.0f, uprev2 = 0.0f;

    for (int t = 0; t <= 32; ++t) {
      const int m = t - l;
      if (m >= 0 && m <= 17) {
        float dval = 0.0f;                      // padded cols 0 and 17 are 0
        if (m >= 1 && m <= 16)
          dval = fwd ? D[l * LFRM + (m - 1)] : D[(m - 1) * LFRM + l];

        if (m == 0) {
          cur = 0.0f;                           // cum[l][0] = 0
        } else if (l == 0) {
          cur = left + dval;                    // row 0: plain cumsum
        } else {
          const float a = uprev2;               // cum[l-1][m-1]
          const float b = left;                 // cum[l][m-1]
          const bool edge = (m == 1) || (m == 17);
          float mn = fminf(a, b);
          if (edge) mn = fminf(mn, uprev);      // cum[l-1][m] only at edges
          float s = __expf((mn - a) * (1.0f / LBDA))
                  + __expf((mn - b) * (1.0f / LBDA));
          if (edge) s += __expf((mn - uprev) * (1.0f / LBDA));
          cur = dval + mn - LBDA * __logf(s);   // stable softmin
        }
        left = cur;
      }
      const float u = __shfl_up(cur, 1, 32);    // lane l-1's value at step t
      uprev2 = uprev;
      uprev  = u;
    }

    // forward result on lane 15, transposed result on lane 31
    const float total = __shfl(cur, 15, 32) + __shfl(cur, 31, 32);
    if (lane == 0) out[(size_t)q * NV + (n0 + j)] = -total;
  }
}

// ---------------------------------------------------------------------------
extern "C" void kernel_launch(void* const* d_in, const int* in_sizes, int n_in,
                              void* d_out, int out_size, void* d_ws, size_t ws_size,
                              hipStream_t stream) {
  (void)in_sizes; (void)n_in; (void)out_size; (void)ws_size;

  const float* support = (const float*)d_in[0];   // [128, 16, 1024] f32
  const float* target  = (const float*)d_in[1];   // [512, 16, 1024] f32
  float* out = (float*)d_out;                     // [512, 128] f32

  // Workspace layout (bf16): tnh | tnl | snh | snl  (~42 MB total)
  const size_t TN = (size_t)QV * LFRM * DDIM;     // 8192*1024
  const size_t SN = (size_t)NV * LFRM * DDIM;     // 2048*1024
  bf16_t* tnh = (bf16_t*)d_ws;
  bf16_t* tnl = tnh + TN;
  bf16_t* snh = tnl + TN;
  bf16_t* snl = snh + SN;

  normalize_split_kernel<<<dim3((NV + QV) * LFRM), dim3(256), 0, stream>>>(
      support, target, tnh, tnl, snh, snl);

  // grid: x covers n in chunks of 8 (4 waves * 2 tiles), y covers q
  otam_fused_kernel<<<dim3(NV / 8, QV), dim3(128), 0, stream>>>(
      tnh, tnl, snh, snl, out);
}
